// A_MultiHeadAttention_46428596470463
// MI455X (gfx1250) — matrix operand order repaired and checked by reading, not compiled, not run on hardware
//
#include <hip/hip_runtime.h>

typedef __bf16 bf16;
typedef __attribute__((ext_vector_type(16))) __bf16 v16bf;
typedef __attribute__((ext_vector_type(8)))  float  v8f;
typedef __attribute__((ext_vector_type(4)))  unsigned int u32x4;
typedef __attribute__((ext_vector_type(8)))  unsigned int u32x8;

// ---------------------------------------------------------------------------
// WMMA helper: D = A(16x32 bf16) x B(32x16 bf16) + C(16x16 f32)
// ---------------------------------------------------------------------------
__device__ __forceinline__ v8f wmma_bf16(v16bf a, v16bf b, v8f c) {
  return __builtin_amdgcn_wmma_f32_16x16x32_bf16(
      /*neg_a=*/false, a, /*neg_b=*/false, b,
      /*c_mod=*/(short)0, c, /*reuse_a=*/false, /*reuse_b=*/false);
}

// ---------------------------------------------------------------------------
// Tensor Data Mover: 2D tile (tile_w elems x tile_h rows, 2B elements) from a
// row-pitched tensor in global memory into LDS, with LDS padding via the D#
// pad fields. Issued by one wave; tracked in-order on that wave's TENSORcnt.
// ---------------------------------------------------------------------------
__device__ __forceinline__ unsigned int lds_addr_of(const void* p) {
  const __attribute__((address_space(3))) char* lp =
      (const __attribute__((address_space(3))) char*)p;
  return (unsigned int)(unsigned long long)lp;
}

__device__ __forceinline__ void tdm_load_2d(const void* gsrc, unsigned lds_off,
                                            unsigned tile_w, unsigned tile_h,
                                            unsigned row_stride_elems,
                                            unsigned pad_en, unsigned pad_int,
                                            unsigned pad_amt) {
  const unsigned long long ga = (unsigned long long)(size_t)gsrc;
  u32x4 g0;
  g0[0] = 1u;                                           // count=1, user mode
  g0[1] = lds_off;                                      // lds_addr (bytes)
  g0[2] = (unsigned)ga;                                 // global_addr[31:0]
  g0[3] = ((unsigned)(ga >> 32) & 0x01FFFFFFu)          // global_addr[56:32]
        | (2u << 30);                                   // type = 2 ("image")
  u32x8 g1;
  g1[0] = (1u << 16)                                    // data_size = 2 bytes
        | (pad_en << 20) | (pad_int << 22) | (pad_amt << 25);
  g1[1] = (row_stride_elems & 0xFFFFu) << 16;           // tensor_dim0[15:0]
  g1[2] = ((row_stride_elems >> 16) & 0xFFFFu)          // tensor_dim0[31:16]
        | ((tile_h & 0xFFFFu) << 16);                   // tensor_dim1[15:0]
  g1[3] = ((tile_h >> 16) & 0xFFFFu)                    // tensor_dim1[31:16]
        | ((tile_w & 0xFFFFu) << 16);                   // tile_dim0
  g1[4] = (tile_h & 0xFFFFu);                           // tile_dim1 (dim2=0)
  g1[5] = row_stride_elems;                             // dim0_stride[31:0]
  g1[6] = 0u;                                           // dim0_stride[47:32]
  g1[7] = 0u;
  const u32x4 z = {0u, 0u, 0u, 0u};                     // 2D: groups 2/3 unused
  asm volatile("tensor_load_to_lds %0, %1, %2, %3"
               :: "s"(g0), "s"(g1), "s"(z), "s"(z)
               : "memory");
}

template <int N>
__device__ __forceinline__ void wait_tensor() {
#if __has_builtin(__builtin_amdgcn_s_wait_tensorcnt)
  __builtin_amdgcn_s_wait_tensorcnt(N);
#else
  asm volatile("s_wait_tensorcnt %0" :: "i"(N) : "memory");
#endif
}

// ---------------------------------------------------------------------------
// xor-shuffle within 16-lane halves via v_permlane16_b32 (VALU, no LDS).
// ---------------------------------------------------------------------------
__device__ __forceinline__ float lane16_xor(float v, int m) {
#if __has_builtin(__builtin_amdgcn_permlane16)
  unsigned s0, s1;
  if (m == 1)      { s0 = 0x67452301u; s1 = 0xEFCDAB89u; }
  else if (m == 2) { s0 = 0x54761032u; s1 = 0xDCFE98BAu; }
  else if (m == 4) { s0 = 0x32107654u; s1 = 0xBA98FEDCu; }
  else             { s0 = 0xFEDCBA98u; s1 = 0x76543210u; }
  unsigned u = __builtin_bit_cast(unsigned, v);
  u = __builtin_amdgcn_permlane16(u, u, s0, s1, false, false);
  return __builtin_bit_cast(float, u);
#else
  return __shfl_xor(v, m, 16);
#endif
}

// ---------------------------------------------------------------------------
// Fragment gathers from LDS, per CDNA5 ISA 7.12.2 16-bit layouts.
// ---------------------------------------------------------------------------
__device__ __forceinline__ v16bf load_a_frag(const bf16* p, int ld, int lane) {
  const int m = lane & 15, h = lane >> 4;
  const bf16* row = p + m * ld;
  v16bf a;
#pragma unroll
  for (int i = 0; i < 4; ++i) {
    const int k = 8 * h + 2 * i;
    a[2 * i]     = row[k];
    a[2 * i + 1] = row[k + 1];
  }
#pragma unroll
  for (int i = 0; i < 4; ++i) {
    const int k = 16 + 8 * h + 2 * i;
    a[8 + 2 * i] = row[k];
    a[9 + 2 * i] = row[k + 1];
  }
  return a;
}

__device__ __forceinline__ v16bf load_b_frag(const bf16* p, int ld, int lane) {
  const int n = lane & 15, h = lane >> 4;
  const bf16* row = p + n * ld;
  v16bf b;
#pragma unroll
  for (int i = 0; i < 8; ++i) {
    const int k = 8 * h + 2 * i + (i >= 4 ? 8 : 0);
    b[2 * i]     = row[k];
    b[2 * i + 1] = row[k + 1];
  }
  return b;
}

__device__ __forceinline__ v16bf load_b_frag_kmajor(const bf16* p, int ld, int lane) {
  const int n = lane & 15, h = lane >> 4;
  v16bf b;
#pragma unroll
  for (int i = 0; i < 8; ++i) {
    const int k = 8 * h + 2 * i + (i >= 4 ? 8 : 0);
    b[2 * i]     = p[k * ld + n];
    b[2 * i + 1] = p[(k + 1) * ld + n];
  }
  return b;
}

// ---------------------------------------------------------------------------
// f32 -> bf16 conversion (grid-stride)
// ---------------------------------------------------------------------------
__global__ void f32_to_bf16_kernel(const float* __restrict__ in,
                                   bf16* __restrict__ out, size_t n) {
  size_t i = (size_t)blockIdx.x * blockDim.x + threadIdx.x;
  const size_t stride = (size_t)gridDim.x * blockDim.x;
  for (; i < n; i += stride) out[i] = (bf16)in[i];
}

// ---------------------------------------------------------------------------
// GEMM: C[m,n] = sum_k A[m,k] * W[n,k] + bias[n]   (torch Linear: x @ W^T + b)
// 128x128 block tile, 4 waves each 64x64 (4x4 WMMA). Double-buffered TDM:
// waves 0/1 own the A/W tile DMA, prefetch next K-step while current computes.
// ---------------------------------------------------------------------------
template <typename OutT>
__global__ __launch_bounds__(128)
void gemm_xwT_kernel(const bf16* __restrict__ A, const bf16* __restrict__ W,
                     const float* __restrict__ bias, OutT* __restrict__ C,
                     int M, int N, int K) {
  __shared__ bf16 As[2][128 * 40];  // 128 rows x 32 k, stride 40 (16B aligned)
  __shared__ bf16 Ws[2][128 * 40];

  const int tid  = threadIdx.x;
  const int lane = tid & 31;
  const int wid  = tid >> 5;
  const int wm   = (wid >> 1) * 64;
  const int wn   = (wid & 1) * 64;
  const int bm   = blockIdx.x * 128;
  const int bn   = blockIdx.y * 128;

  const unsigned As_off[2] = {lds_addr_of(As[0]), lds_addr_of(As[1])};
  const unsigned Ws_off[2] = {lds_addr_of(Ws[0]), lds_addr_of(Ws[1])};

  const bf16* gA = A + (size_t)bm * K;
  const bf16* gW = W + (size_t)bn * K;

  v8f acc[4][4] = {};

  // Prologue: stage K-step 0 into buffer 0.
  // Row = 32 elems = 16 DWORDs (pad_int 3), pad 4 DWORDs (pad_amt 3) -> ld 40.
  if (wid == 0)      tdm_load_2d(gA, As_off[0], 32, 128, (unsigned)K, 1, 3, 3);
  else if (wid == 1) tdm_load_2d(gW, Ws_off[0], 32, 128, (unsigned)K, 1, 3, 3);

  const int steps = K >> 5;
  for (int i = 0; i < steps; ++i) {
    const int cur = i & 1;
    // Prefetch next K-step into the other buffer, then wait for the current
    // buffer's (older, in-order) TDM only: tensorcnt <= 1 leaves the prefetch
    // in flight behind the WMMA phase.
    if (wid == 0) {
      if (i + 1 < steps) {
        tdm_load_2d(gA + (i + 1) * 32, As_off[1 - cur], 32, 128, (unsigned)K, 1, 3, 3);
        wait_tensor<1>();
      } else {
        wait_tensor<0>();
      }
    } else if (wid == 1) {
      if (i + 1 < steps) {
        tdm_load_2d(gW + (i + 1) * 32, Ws_off[1 - cur], 32, 128, (unsigned)K, 1, 3, 3);
        wait_tensor<1>();
      } else {
        wait_tensor<0>();
      }
    }
    __syncthreads();

    v16bf af[4], wf[4];
#pragma unroll
    for (int t = 0; t < 4; ++t)
      af[t] = load_a_frag(As[cur] + (wm + t * 16) * 40, 40, lane);
#pragma unroll
    for (int t = 0; t < 4; ++t)
      wf[t] = load_b_frag(Ws[cur] + (wn + t * 16) * 40, 40, lane);

#pragma unroll
    for (int mt = 0; mt < 4; ++mt)
#pragma unroll
      for (int nt = 0; nt < 4; ++nt)
        acc[mt][nt] = wmma_bf16(af[mt], wf[nt], acc[mt][nt]);
    __syncthreads();  // readers done before buffer is overwritten next iter
  }

  const int h = lane >> 4, nloc = lane & 15;
#pragma unroll
  for (int mt = 0; mt < 4; ++mt)
#pragma unroll
    for (int nt = 0; nt < 4; ++nt)
#pragma unroll
      for (int r = 0; r < 8; ++r) {
        const int row = bm + wm + mt * 16 + r + 8 * h;
        const int col = bn + wn + nt * 16 + nloc;
        const float v = acc[mt][nt][r] + bias[col];
        C[(size_t)row * N + col] = (OutT)v;
      }
}

// ---------------------------------------------------------------------------
// Flash attention over UNSCALED logits (faithful to reference).
// qkv: bf16 [B, S, 3E]; head hd channels: q=hd*192, k=+64, v=+128.
// vals: bf16 [B, S, E], channel = hd*64 + d.
// grid: (B*H, S/64); 128 threads = 4 waves; wave = 16 queries x Dh(64).
// Double-buffered TDM staging of K/V tiles (wave 0 owns the DMA).
// ---------------------------------------------------------------------------
__global__ __launch_bounds__(128)
void attn_kernel(const bf16* __restrict__ qkv, bf16* __restrict__ vals) {
  constexpr int S  = 1024;
  constexpr int E  = 1024;
  constexpr int E3 = 3072;

  const int bh  = blockIdx.x;
  const int b   = bh >> 4;
  const int hd  = bh & 15;
  const int tid = threadIdx.x, lane = tid & 31, wid = tid >> 5;
  const size_t tokbase = (size_t)b * S;
  const int qch = hd * 192;
  const int kch = hd * 192 + 64;
  const int vch = hd * 192 + 128;

  __shared__ bf16 Ks[2][32 * 72];  // 32 keys x 64 dims, padded stride 72
  __shared__ bf16 Vs[2][32 * 72];
  __shared__ bf16 Ps[4][16 * 40];  // per-wave P transpose staging

  const unsigned Ks_off[2] = {lds_addr_of(Ks[0]), lds_addr_of(Ks[1])};
  const unsigned Vs_off[2] = {lds_addr_of(Vs[0]), lds_addr_of(Vs[1])};

  // Per-wave Q fragments: 16 queries x 64 dims = 2 A-frags (K=32 each).
  const int q0 = blockIdx.y * 64 + wid * 16;
  v16bf qa[2];
  {
    const int m = lane & 15, h = lane >> 4;
    const bf16* qrow = qkv + (tokbase + q0 + m) * E3 + qch;
#pragma unroll
    for (int half = 0; half < 2; ++half) {
      v16bf a;
#pragma unroll
      for (int i = 0; i < 4; ++i) {
        const int k = half * 32 + 8 * h + 2 * i;
        a[2 * i]     = qrow[k];
        a[2 * i + 1] = qrow[k + 1];
      }
#pragma unroll
      for (int i = 0; i < 4; ++i) {
        const int k = half * 32 + 16 + 8 * h + 2 * i;
        a[8 + 2 * i] = qrow[k];
        a[9 + 2 * i] = qrow[k + 1];
      }
      qa[half] = a;
    }
  }

  v8f o[4] = {};
  float mrow[8], lrow[8];
#pragma unroll
  for (int r = 0; r < 8; ++r) { mrow[r] = -3.0e38f; lrow[r] = 0.0f; }

  // Prologue: stage key-block 0 into buffer 0.
  // Row = 64 elems = 32 DWORDs (pad_int 4), pad 4 DWORDs (pad_amt 3) -> ld 72.
  if (wid == 0) {
    tdm_load_2d(qkv + tokbase * E3 + kch, Ks_off[0], 64, 32, E3, 1, 4, 3);
    tdm_load_2d(qkv + tokbase * E3 + vch, Vs_off[0], 64, 32, E3, 1, 4, 3);
  }

  constexpr int steps = S / 32;
  for (int i = 0; i < steps; ++i) {
    const int cur = i & 1;
    // Prefetch next K/V pair; wait only for the current pair (older 2 of the
    // wave's in-order TDM ops): tensorcnt <= 2.
    if (wid == 0) {
      if (i + 1 < steps) {
        const size_t nrow = (tokbase + (i + 1) * 32) * E3;
        tdm_load_2d(qkv + nrow + kch, Ks_off[1 - cur], 64, 32, E3, 1, 4, 3);
        tdm_load_2d(qkv + nrow + vch, Vs_off[1 - cur], 64, 32, E3, 1, 4, 3);
        wait_tensor<2>();
      } else {
        wait_tensor<0>();
      }
    }
    __syncthreads();

    // Logits: two 16-key groups, Dh=64 -> 2 chained WMMAs per group.
    v8f st[2];
#pragma unroll
    for (int g = 0; g < 2; ++g) {
      const v16bf kb0 = load_b_frag(Ks[cur] + (g * 16) * 72, 72, lane);
      const v16bf kb1 = load_b_frag(Ks[cur] + (g * 16) * 72 + 32, 72, lane);
      v8f c = {};
      c = wmma_bf16(qa[0], kb0, c);
      c = wmma_bf16(qa[1], kb1, c);
      st[g] = c;
    }

    // Online softmax; row (r + 8*(lane>>4)) lives across 16 lanes.
#pragma unroll
    for (int r = 0; r < 8; ++r) {
      float mx = fmaxf(st[0][r], st[1][r]);
      mx = fmaxf(mx, lane16_xor(mx, 8));
      mx = fmaxf(mx, lane16_xor(mx, 4));
      mx = fmaxf(mx, lane16_xor(mx, 2));
      mx = fmaxf(mx, lane16_xor(mx, 1));
      const float mnew  = fmaxf(mrow[r], mx);
      const float scale = __expf(mrow[r] - mnew);
      const float p0 = __expf(st[0][r] - mnew);
      const float p1 = __expf(st[1][r] - mnew);
      float rs = p0 + p1;
      rs += lane16_xor(rs, 8);
      rs += lane16_xor(rs, 4);
      rs += lane16_xor(rs, 2);
      rs += lane16_xor(rs, 1);
      lrow[r] = lrow[r] * scale + rs;
      mrow[r] = mnew;
      st[0][r] = p0;
      st[1][r] = p1;
#pragma unroll
      for (int t = 0; t < 4; ++t) o[t][r] *= scale;
    }

    // Transpose P: C-layout -> row-major LDS (16 x 32 bf16), reload as A-frag.
    {
      const int h = lane >> 4, n = lane & 15;
      bf16* ps = Ps[wid];
#pragma unroll
      for (int g = 0; g < 2; ++g)
#pragma unroll
        for (int r = 0; r < 8; ++r)
          ps[(r + 8 * h) * 40 + g * 16 + n] = (bf16)st[g][r];
    }
    // Per-wave LDS RAW: DS ops in-order per wave; wait for stores to land.
    asm volatile("s_wait_dscnt 0" ::: "memory");
    const v16bf pa = load_a_frag(Ps[wid], 40, lane);

    // PV: P(16x32) x V(32x64) -> 4 N-tiles; V read K-major from LDS.
#pragma unroll
    for (int t = 0; t < 4; ++t) {
      const v16bf vb = load_b_frag_kmajor(Vs[cur] + t * 16, 72, lane);
      o[t] = wmma_bf16(pa, vb, o[t]);
    }
    __syncthreads();  // readers done before buffer is overwritten next iter
  }

  // Epilogue: normalize by row sum, store bf16 into vals [B,S,E].
  {
    const int h = lane >> 4, n = lane & 15;
#pragma unroll
    for (int r = 0; r < 8; ++r) {
      const float inv = 1.0f / lrow[r];
      const int tok = q0 + r + 8 * h;
#pragma unroll
      for (int t = 0; t < 4; ++t) {
        const int c = hd * 64 + t * 16 + n;
        vals[(tokbase + tok) * E + c] = (bf16)(o[t][r] * inv);
      }
    }
  }
}

// ---------------------------------------------------------------------------
// Launcher
// ---------------------------------------------------------------------------
extern "C" void kernel_launch(void* const* d_in, const int* in_sizes, int n_in,
                              void* d_out, int out_size, void* d_ws, size_t ws_size,
                              hipStream_t stream) {
  (void)in_sizes; (void)n_in; (void)out_size; (void)ws_size;

  const float* x     = (const float*)d_in[0];  // [8,1024,1024]
  const float* qkv_w = (const float*)d_in[1];  // [3072,1024]
  const float* qkv_b = (const float*)d_in[2];  // [3072]
  const float* o_w   = (const float*)d_in[3];  // [1024,1024]
  const float* o_b   = (const float*)d_in[4];  // [1024]
  float* out = (float*)d_out;                  // [8,1024,1024]

  constexpr size_t NTOK = 8 * 1024;   // B*S
  constexpr size_t D    = 1024;
  constexpr size_t E    = 1024;
  constexpr size_t E3   = 3072;

  char* ws = (char*)d_ws;
  bf16* xb   = (bf16*)ws; ws += NTOK * D * sizeof(bf16);   // 16 MB
  bf16* wqkv = (bf16*)ws; ws += E3 * D * sizeof(bf16);     //  6 MB
  bf16* wo   = (bf16*)ws; ws += E * E * sizeof(bf16);      //  2 MB
  bf16* qkv  = (bf16*)ws; ws += NTOK * E3 * sizeof(bf16);  // 48 MB
  bf16* vals = (bf16*)ws;                                  // 16 MB

  // 1) downconvert inputs to bf16
  f32_to_bf16_kernel<<<2048, 256, 0, stream>>>(x, xb, NTOK * D);
  f32_to_bf16_kernel<<<1024, 256, 0, stream>>>(qkv_w, wqkv, E3 * D);
  f32_to_bf16_kernel<<<512, 256, 0, stream>>>(o_w, wo, E * E);

  // 2) QKV projection: [8192,1024] x [3072,1024]^T -> bf16 [8192,3072]
  gemm_xwT_kernel<bf16><<<dim3(NTOK / 128, E3 / 128), 128, 0, stream>>>(
      xb, wqkv, qkv_b, qkv, (int)NTOK, (int)E3, (int)D);

  // 3) attention per (b,h), 64 queries per block
  attn_kernel<<<dim3(8 * 16, 1024 / 64), 128, 0, stream>>>(qkv, vals);

  // 4) output projection: [8192,1024] x [1024,1024]^T + b -> f32 out
  gemm_xwT_kernel<float><<<dim3(NTOK / 128, E / 128), 128, 0, stream>>>(
      vals, wo, o_b, out, (int)NTOK, (int)E, (int)E);
}
